// SimpleGAT_86002425135778
// MI455X (gfx1250) — matrix operand order repaired and checked by reading, not compiled, not run on hardware
//
#include <hip/hip_runtime.h>

typedef __attribute__((ext_vector_type(16))) __bf16 v16bf;
typedef __attribute__((ext_vector_type(8)))  float  v8f;

#define N_NODES 8192
#define F_IN    512
#define F_OUT   256
#define NEG_INF -1000000000.0f

union Frag16 {
    v16bf v;
    unsigned short u[16];
    unsigned int   d[8];
};

// f32 -> bf16 bits, round-to-nearest-even
static __device__ __forceinline__ unsigned short f2bf(float f) {
    union { float f; unsigned int u; } x; x.f = f;
    unsigned int r = x.u + 0x7FFFu + ((x.u >> 16) & 1u);
    return (unsigned short)(r >> 16);
}

// ---------------------------------------------------------------- K1: convert
__global__ void cvt_bf16_kernel(const float* __restrict__ src,
                                unsigned short* __restrict__ dst, int n) {
    int i = blockIdx.x * blockDim.x + threadIdx.x;
    if (i < n) dst[i] = f2bf(src[i]);
}

// ---------------------------------------------------------------- K2: h = X @ W (bf16 WMMA)
// One wave computes one 16x16 tile of H. K-loop: F_IN/32 = 16 WMMAs.
__global__ __launch_bounds__(128)
void gemm1_wmma_kernel(const unsigned short* __restrict__ Abf,  // [N, F_IN] bf16
                       const unsigned short* __restrict__ Bbf,  // [F_IN, F_OUT] bf16
                       float* __restrict__ H,                   // [N, F_OUT] f32
                       unsigned short* __restrict__ Hbf)        // [N, F_OUT] bf16
{
    const int wave = threadIdx.x >> 5;          // 4 waves / block
    const int lane = threadIdx.x & 31;
    const int tile = blockIdx.x * 4 + wave;
    const int tiles_n = F_OUT / 16;             // 16
    const int tm = tile / tiles_n;
    const int tn = tile % tiles_n;
    const int row0 = tm * 16, col0 = tn * 16;
    const int m = lane & 15, half = lane >> 4, n = m;

    v8f acc = {};
    for (int k0 = 0; k0 < F_IN; k0 += 32) {
        Frag16 fa, fb;
        #pragma unroll
        for (int v = 0; v < 8; ++v) {
            // A (16x32 bf16): lane half selects K block; VGPR v -> K pair
            int kb = 2 * v + ((v >> 2) << 3) + 8 * half;
            fa.d[v] = *(const unsigned int*)(Abf + (size_t)(row0 + m) * F_IN + k0 + kb);
            // B (32x16 bf16): rows k=2v+8*(v>>2)+8*half, k+1; column col0+n
            int kr = k0 + 2 * v + ((v >> 2) << 3) + 8 * half;
            unsigned int lo = Bbf[(size_t)kr       * F_OUT + col0 + n];
            unsigned int hi = Bbf[(size_t)(kr + 1) * F_OUT + col0 + n];
            fb.d[v] = lo | (hi << 16);
        }
        acc = __builtin_amdgcn_wmma_f32_16x16x32_bf16(
                  false, fa.v, false, fb.v, (short)0, acc, false, false);
    }
    #pragma unroll
    for (int r = 0; r < 8; ++r) {
        int gm = row0 + r + 8 * half;
        float val = acc[r];
        H  [(size_t)gm * F_OUT + col0 + n] = val;
        Hbf[(size_t)gm * F_OUT + col0 + n] = f2bf(val);
    }
}

// ---------------------------------------------------------------- K3: hs = h@a_src, hd = h@a_dst
__global__ __launch_bounds__(256)
void gemv_attn_kernel(const float* __restrict__ H, const float* __restrict__ a,
                      float* __restrict__ hs, float* __restrict__ hd)
{
    int row  = blockIdx.x * 8 + (threadIdx.x >> 5);
    int lane = threadIdx.x & 31;
    const float* hrow = H + (size_t)row * F_OUT;
    float s = 0.f, d = 0.f;
    #pragma unroll
    for (int c = lane; c < F_OUT; c += 32) {
        float h = hrow[c];
        s += h * a[c];
        d += h * a[F_OUT + c];
    }
    #pragma unroll
    for (int off = 16; off; off >>= 1) {
        s += __shfl_xor(s, off, 32);
        d += __shfl_xor(d, off, 32);
    }
    if (lane == 0) { hs[row] = s; hd[row] = d; }
}

// ---------------------------------------------------------------- K4: per-row softmax stats
// One block per row; masked-e staged in 32KB LDS so adj row is read exactly once.
// adj loads are non-temporal (streaming, 256MB) to keep Hbf/hd resident in L2.
__global__ __launch_bounds__(256)
void softmax_stats_kernel(const float* __restrict__ adj, const float* __restrict__ hs,
                          const float* __restrict__ hd, float* __restrict__ rmax,
                          float* __restrict__ rsum)
{
    __shared__ float e_s[N_NODES];   // 32 KB (LDS is 320 KB/WGP on CDNA5)
    __shared__ float red[256];
    const int i = blockIdx.x;
    const int t = threadIdx.x;
    const float his = hs[i];
    const float* arow = adj + (size_t)i * N_NODES;

    float mx = -3.4e38f;
    for (int j = t; j < N_NODES; j += 256) {
        float x = his + hd[j];
        float e = (x > 0.f) ? x : 0.01f * x;       // leaky_relu(0.01)
        e += NEG_INF * (1.0f - __builtin_nontemporal_load(arow + j));
        e_s[j] = e;
        mx = fmaxf(mx, e);
    }
    red[t] = mx; __syncthreads();
    #pragma unroll
    for (int off = 128; off; off >>= 1) {
        if (t < off) red[t] = fmaxf(red[t], red[t + off]);
        __syncthreads();
    }
    mx = red[0]; __syncthreads();

    float sm = 0.f;
    for (int j = t; j < N_NODES; j += 256) sm += __expf(e_s[j] - mx);
    red[t] = sm; __syncthreads();
    #pragma unroll
    for (int off = 128; off; off >>= 1) {
        if (t < off) red[t] += red[t + off];
        __syncthreads();
    }
    if (t == 0) { rmax[i] = mx; rsum[i] = red[0]; }
}

// ---------------------------------------------------------------- K5: fused attention write + out = attn @ h
// Block = 8 waves = 16-row strip. The 16x32 adj tile for the NEXT chunk is pulled into
// a double-buffered LDS slot with GLOBAL_LOAD_ASYNC_TO_LDS_B128 (ASYNCcnt) while the
// current chunk computes probabilities + WMMAs, hiding the 256MB stream behind compute.
// Attention stores are non-temporal (write-once, never re-read).
__global__ __launch_bounds__(256)
void attn_out_kernel(const float* __restrict__ adj,
                     const float* __restrict__ hs, const float* __restrict__ hd,
                     const float* __restrict__ rmax, const float* __restrict__ rsum,
                     const unsigned short* __restrict__ Hbf,
                     float* __restrict__ out, float* __restrict__ att)
{
    __shared__ float ldsAdj[2][16][32];         // 2 x 2KB adj tiles, double-buffered
    __shared__ unsigned short ldsA[16][36];     // 16x32 bf16 attn tile, padded rows

    const int i0 = blockIdx.x * 16;
    const int t  = threadIdx.x;
    const int r  = t >> 4;                      // attn-tile row 0..15
    const int c  = t & 15;                      // attn-tile col 0..15
    const float his = hs[i0 + r];
    const float mr  = rmax[i0 + r];
    const float isr = 1.0f / rsum[i0 + r];

    const int wave = t >> 5;                    // 0..7 -> owns F_out tiles 2w, 2w+1
    const int lane = t & 31;
    const int m = lane & 15, half = lane >> 4, n = m;
    const int c0 = (2 * wave) * 16;
    const int c1 = (2 * wave + 1) * 16;

    // async-copy lane assignment: threads 0..127 (waves 0-3, uniform EXEC), 16B each
    const int ar = t >> 3;                      // tile row 0..15
    const int aq = t & 7;                       // 16B chunk 0..7 within row

    auto issue_adj_async = [&](int j0n, int p) {
        if (t < 128) {
            unsigned lds_off = (unsigned)(size_t)&ldsAdj[p][ar][aq * 4];
            unsigned long long gaddr =
                (unsigned long long)(size_t)(adj + (size_t)(i0 + ar) * N_NODES + j0n + aq * 4);
            asm volatile("global_load_async_to_lds_b128 %0, %1, off"
                         :: "v"(lds_off), "v"(gaddr) : "memory");
        }
    };

    issue_adj_async(0, 0);                      // prologue: tile 0 -> buffer 0

    v8f acc0 = {}, acc1 = {};

    for (int j0 = 0; j0 < N_NODES; j0 += 32) {
        const int p = (j0 >> 5) & 1;
        asm volatile("s_wait_asynccnt 0x0" ::: "memory");
        __syncthreads();                        // tile p visible to all waves
        if (j0 + 32 < N_NODES) issue_adj_async(j0 + 32, p ^ 1);

        // ---- attention probabilities for this 16x32 tile (2 elems / thread)
        #pragma unroll
        for (int s = 0; s < 2; ++s) {
            int j = j0 + c + 16 * s;
            float x = his + hd[j];
            float e = (x > 0.f) ? x : 0.01f * x;
            e += NEG_INF * (1.0f - ldsAdj[p][r][c + 16 * s]);
            float pr = __expf(e - mr) * isr;
            __builtin_nontemporal_store(pr, att + (size_t)(i0 + r) * N_NODES + j);
            ldsA[r][c + 16 * s] = f2bf(pr);
        }
        __syncthreads();                        // ldsA tile ready

        // ---- A fragment (same attn tile for all 8 waves), from LDS
        Frag16 fa;
        #pragma unroll
        for (int v = 0; v < 8; ++v) {
            int kb = 2 * v + ((v >> 2) << 3) + 8 * half;
            fa.d[v] = *(const unsigned int*)&ldsA[m][kb];
        }
        // ---- B fragments: Hbf rows j0..j0+31, this wave's two 16-col tiles (L2-resident)
        Frag16 fb0, fb1;
        #pragma unroll
        for (int v = 0; v < 8; ++v) {
            int kr = j0 + 2 * v + ((v >> 2) << 3) + 8 * half;
            unsigned int lo0 = Hbf[(size_t)kr       * F_OUT + c0 + n];
            unsigned int hi0 = Hbf[(size_t)(kr + 1) * F_OUT + c0 + n];
            unsigned int lo1 = Hbf[(size_t)kr       * F_OUT + c1 + n];
            unsigned int hi1 = Hbf[(size_t)(kr + 1) * F_OUT + c1 + n];
            fb0.d[v] = lo0 | (hi0 << 16);
            fb1.d[v] = lo1 | (hi1 << 16);
        }
        acc0 = __builtin_amdgcn_wmma_f32_16x16x32_bf16(
                   false, fa.v, false, fb0.v, (short)0, acc0, false, false);
        acc1 = __builtin_amdgcn_wmma_f32_16x16x32_bf16(
                   false, fa.v, false, fb1.v, (short)0, acc1, false, false);
        __syncthreads();                        // done with ldsA before next overwrite
    }

    #pragma unroll
    for (int rr = 0; rr < 8; ++rr) {
        int gm = i0 + rr + 8 * half;
        out[(size_t)gm * F_OUT + c0 + n] = acc0[rr];
        out[(size_t)gm * F_OUT + c1 + n] = acc1[rr];
    }
}

// ---------------------------------------------------------------- launch
extern "C" void kernel_launch(void* const* d_in, const int* in_sizes, int n_in,
                              void* d_out, int out_size, void* d_ws, size_t ws_size,
                              hipStream_t stream) {
    const float* features = (const float*)d_in[0];   // [8192, 512]
    const float* adj      = (const float*)d_in[1];   // [8192, 8192]
    const float* W        = (const float*)d_in[2];   // [512, 256]
    const float* a        = (const float*)d_in[3];   // [512]

    float* out = (float*)d_out;                      // [8192, 256]
    float* att = out + (size_t)N_NODES * F_OUT;      // [8192, 8192]

    char* ws = (char*)d_ws;
    unsigned short* featbf = (unsigned short*)ws; ws += (size_t)N_NODES * F_IN * 2;
    unsigned short* Wbf    = (unsigned short*)ws; ws += (size_t)F_IN * F_OUT * 2;
    float*          H      = (float*)ws;          ws += (size_t)N_NODES * F_OUT * 4;
    unsigned short* Hbf    = (unsigned short*)ws; ws += (size_t)N_NODES * F_OUT * 2;
    float*          hs     = (float*)ws;          ws += (size_t)N_NODES * 4;
    float*          hd     = (float*)ws;          ws += (size_t)N_NODES * 4;
    float*          rmax   = (float*)ws;          ws += (size_t)N_NODES * 4;
    float*          rsum   = (float*)ws;          ws += (size_t)N_NODES * 4;

    cvt_bf16_kernel<<<(N_NODES * F_IN) / 256, 256, 0, stream>>>(features, featbf, N_NODES * F_IN);
    cvt_bf16_kernel<<<(F_IN * F_OUT) / 256, 256, 0, stream>>>(W, Wbf, F_IN * F_OUT);

    // 512 row-tiles x 16 col-tiles = 8192 tiles, 4 waves/block
    gemm1_wmma_kernel<<<2048, 128, 0, stream>>>(featbf, Wbf, H, Hbf);

    gemv_attn_kernel<<<N_NODES / 8, 256, 0, stream>>>(H, a, hs, hd);

    softmax_stats_kernel<<<N_NODES, 256, 0, stream>>>(adj, hs, hd, rmax, rsum);

    attn_out_kernel<<<N_NODES / 16, 256, 0, stream>>>(adj, hs, hd, rmax, rsum, Hbf, out, att);
}
